// S4Att_84997402787904
// MI455X (gfx1250) — compile-verified, matrix operand order checked by reading, not run on hardware
//
#include <hip/hip_runtime.h>
#include <hip/hip_bf16.h>

// ---------------------------------------------------------------------------
// S4D stack for MI455X (gfx1250, wave32, WMMA).
// Chunked-scan SSM conv (T=64) + GLU GEMM on v_wmma_f32_16x16x32_f16.
// LDS fragment traffic: 16B ds_load_b128. Untransformed staging copies use
// CDNA5 async global->LDS DMA (global_load_async_to_lds_b128 / ASYNCcnt).
// ---------------------------------------------------------------------------

#define B_  16
#define L_  2048
#define H_  256
#define N_  64
#define NL_ 4
#define T_  64               // chunk length
#define C_  32               // L_/T_ chunks
#define H2_ 512              // 2H

typedef __attribute__((ext_vector_type(16))) _Float16 v16h;
typedef __attribute__((ext_vector_type(8)))  _Float16 v8h;
typedef __attribute__((ext_vector_type(8)))  float    v8f;
typedef __attribute__((ext_vector_type(4)))  float    v4f;

// -------------------- CDNA5 async global->LDS (16B/lane) -------------------
// ISA 15.18.3: GLOBAL_LOAD_ASYNC_TO_LDS_B128, tracked with ASYNCcnt.
// LDS offset = low 32 bits of the generic pointer (aperture: addr[31:0]).
__device__ __forceinline__ void async_copy16(void* ldst, const void* gsrc) {
  const unsigned lds = (unsigned)(uintptr_t)ldst;
  const unsigned long long ga = (unsigned long long)(uintptr_t)gsrc;
  asm volatile("global_load_async_to_lds_b128 %0, %1, off"
               :: "v"(lds), "v"(ga) : "memory");
}
__device__ __forceinline__ void async_wait0() {
  asm volatile("s_wait_asynccnt 0" ::: "memory");
}

// ------------------------- WMMA fragment helpers ---------------------------
// A fragment (16x32 MxK, f16): per lane two contiguous 8-half runs at
// k = kb..kb+7 and k = 16+kb..16+kb+7  (kb = 0 or 8).  -> 2x ds_load_b128.
__device__ __forceinline__ v16h frag_a(const _Float16* base, int row0, int col0, int ld) {
  const int lane = threadIdx.x & 31;
  const int m  = lane & 15;
  const int kb = (lane & 16) ? 8 : 0;
  const _Float16* p = base + (row0 + m) * ld + col0 + kb;
  const v8h lo = *(const v8h*)p;
  const v8h hi = *(const v8h*)(p + 16);
  return __builtin_shufflevector(lo, hi, 0, 1, 2, 3, 4, 5, 6, 7,
                                         8, 9, 10, 11, 12, 13, 14, 15);
}
// B fragment (32x16 KxN, f16) read from an N-major (transposed) LDS tile
// laid out [N][K] with row stride ldk (multiple of 8): same contiguity trick.
__device__ __forceinline__ v16h frag_bT(const _Float16* base, int k0, int n0, int ldk) {
  const int lane = threadIdx.x & 31;
  const int n  = lane & 15;
  const int kb = (lane & 16) ? 8 : 0;
  const _Float16* p = base + (n0 + n) * ldk + k0 + kb;
  const v8h lo = *(const v8h*)p;
  const v8h hi = *(const v8h*)(p + 16);
  return __builtin_shufflevector(lo, hi, 0, 1, 2, 3, 4, 5, 6, 7,
                                         8, 9, 10, 11, 12, 13, 14, 15);
}
__device__ __forceinline__ v8f wmma_f16(v16h a, v16h b, v8f c) {
  return __builtin_amdgcn_wmma_f32_16x16x32_f16(false, a, false, b, (short)0, c, false, false);
}
// C/D 16x16 f32: lane n = lane&15, VGPR r -> m = r + (lane&16 ? 8 : 0)
__device__ __forceinline__ void d_lane(int& mb, int& n) {
  const int lane = threadIdx.x & 31;
  n  = lane & 15;
  mb = (lane & 16) ? 8 : 0;
}
__device__ __forceinline__ v8h cvt8(v4f a, v4f b) {
  v8h r;
#pragma unroll
  for (int e = 0; e < 4; ++e) { r[e] = (_Float16)a[e]; r[4 + e] = (_Float16)b[e]; }
  return r;
}

__device__ __forceinline__ float gelu_f(float x) {
  const float k0 = 0.7978845608028654f, k1 = 0.044715f;
  return 0.5f * x * (1.0f + tanhf(k0 * (x + k1 * x * x * x)));
}
__device__ __forceinline__ float sigmoid_f(float x) { return 1.0f / (1.0f + expf(-x)); }

// --------------------------- transposes ------------------------------------
__global__ void k_tr_LH_to_HL(const float* __restrict__ in, float* __restrict__ out) {
  __shared__ float tile[32][33];
  const int b = blockIdx.z, h0 = blockIdx.x * 32, l0 = blockIdx.y * 32;
  for (int r = threadIdx.y; r < 32; r += 8)
    tile[r][threadIdx.x] = in[((size_t)b * L_ + l0 + r) * H_ + h0 + threadIdx.x];
  __syncthreads();
  for (int r = threadIdx.y; r < 32; r += 8)
    out[((size_t)b * H_ + h0 + r) * L_ + l0 + threadIdx.x] = tile[threadIdx.x][r];
}
__global__ void k_tr_HL_to_LH(const float* __restrict__ in, float* __restrict__ out) {
  __shared__ float tile[32][33];
  const int b = blockIdx.z, l0 = blockIdx.x * 32, h0 = blockIdx.y * 32;
  for (int r = threadIdx.y; r < 32; r += 8)
    tile[r][threadIdx.x] = in[((size_t)b * H_ + h0 + r) * L_ + l0 + threadIdx.x];
  __syncthreads();
  for (int r = threadIdx.y; r < 32; r += 8)
    out[((size_t)b * L_ + l0 + r) * H_ + h0 + threadIdx.x] = tile[threadIdx.x][r];
}

// --------------------------- LayerNorm over H ------------------------------
__global__ void k_layernorm(const float* __restrict__ Hres,
                            const float* __restrict__ scale,
                            const float* __restrict__ bias,
                            _Float16* __restrict__ u16) {
  const int gid = blockIdx.x * 256 + threadIdx.x;
  const int b = gid / L_, l = gid % L_;
  const float* p = Hres + (size_t)b * H_ * L_ + l;
  float s = 0.f, s2 = 0.f;
  for (int h = 0; h < H_; ++h) {
    const float v = p[(size_t)h * L_];
    s += v; s2 += v * v;
  }
  const float mu  = s * (1.0f / H_);
  const float var = s2 * (1.0f / H_) - mu * mu;
  const float inv = rsqrtf(var + 1e-5f);
  _Float16* q = u16 + (size_t)b * H_ * L_ + l;
  for (int h = 0; h < H_; ++h) {
    const float v = p[(size_t)h * L_];
    q[(size_t)h * L_] = (_Float16)(((v - mu) * inv) * scale[h] + bias[h]);
  }
}

// ---------------- per-layer SSM matrices (Vandermonde family) --------------
__global__ void k_build(const float* __restrict__ log_dt, const float* __restrict__ Alog,
                        const float* __restrict__ Aim,    const float* __restrict__ Cre,
                        const float* __restrict__ Cim,
                        float* __restrict__ wCd,
                        _Float16* __restrict__ Ktri,   // (H,64,64)
                        _Float16* __restrict__ Wst,    // (H,128,64)
                        _Float16* __restrict__ G) {    // (H,64,128)
  const int h = blockIdx.x, t = threadIdx.x;
  __shared__ float sAr[N_], sAi[N_], sCr[N_], sCi[N_], sK[T_];
  { // phase 1: thread t == state n computes discretized params
    const int n = t;
    const float dt  = expf(log_dt[h]);
    const float Are = -expf(Alog[h * N_ + n]);
    const float Aimv = Aim[h * N_ + n];
    const float ar = Are * dt, ai = Aimv * dt;          // dtA
    const float ew = expf(ar);
    const float wr = ew * cosf(ai), wi = ew * sinf(ai); // w = exp(dtA)
    const float cr = Cre[h * N_ + n], ci = Cim[h * N_ + n];
    const float nr = wr - 1.f, ni = wi;                 // (exp(dtA)-1)
    const float tr = cr * nr - ci * ni, ti = cr * ni + ci * nr;
    const float den = Are * Are + Aimv * Aimv;
    const float cdr = (tr * Are + ti * Aimv) / den;     // C*(w-1)/A
    const float cdi = (ti * Are - tr * Aimv) / den;
    sAr[n] = ar; sAi[n] = ai; sCr[n] = cdr; sCi[n] = cdi;
    float* wp = wCd + ((size_t)h * N_ + n) * 4;
    wp[0] = ar; wp[1] = ai; wp[2] = cdr; wp[3] = cdi;
  }
  __syncthreads();
  // phase 2: thread t == time index
  float Kt = 0.f;
  for (int n = 0; n < N_; ++n) {
    const float ar = sAr[n], ai = sAi[n], cdr = sCr[n], cdi = sCi[n];
    {
      const float e = expf((float)t * ar);
      const float pr = e * cosf((float)t * ai), pi = e * sinf((float)t * ai);
      Kt += 2.f * (cdr * pr - cdi * pi);                 // K[t] = 2 Re(Cd w^t)
    }
    {
      const float p = (float)(T_ - 1 - t);               // Wst col t: w^{T-1-t}
      const float e = expf(p * ar);
      Wst[((size_t)h * 2 * N_ + n) * T_ + t]      = (_Float16)(e * cosf(p * ai));
      Wst[((size_t)h * 2 * N_ + N_ + n) * T_ + t] = (_Float16)(e * sinf(p * ai));
    }
    {
      const float p = (float)(t + 1);                    // G row t: 2 Cd w^{t+1}
      const float e = expf(p * ar);
      const float pr = e * cosf(p * ai), pi = e * sinf(p * ai);
      G[((size_t)h * T_ + t) * (2 * N_) + n]      = (_Float16)(2.f * (cdr * pr - cdi * pi));
      G[((size_t)h * T_ + t) * (2 * N_) + N_ + n] = (_Float16)(-2.f * (cdr * pi + cdi * pr));
    }
  }
  sK[t] = Kt;
  __syncthreads();
  for (int j = 0; j < T_; ++j)                           // lower-tri Toeplitz
    Ktri[((size_t)h * T_ + t) * T_ + j] = (j <= t) ? (_Float16)sK[t - j] : (_Float16)0.0f;
}

// ---------------- chunk-state projection: P = Wst[h] @ U ------------------
// grid (H_, 4), block 256 (8 waves). S: (B,H,C,128) f32, col = b*32 + c.
#define LDU_ 72                                          // UtT row stride (pad, mult of 8)
__global__ void k_chunk_states(const _Float16* __restrict__ u16,
                               const _Float16* __restrict__ Wst,
                               float* __restrict__ S) {
  const int h = blockIdx.x;
  __shared__ alignas(16) _Float16 Wl[128 * 64];
  __shared__ alignas(16) _Float16 UtT[32 * LDU_];        // [col][t]
#pragma unroll
  for (int i = 0; i < 4; ++i) {                          // 8192 halfs via async DMA
    const int o = threadIdx.x * 8 + 2048 * i;
    async_copy16(&Wl[o], &Wst[(size_t)h * 128 * 64 + o]);
  }
  async_wait0();
  __syncthreads();
  const int w = threadIdx.x >> 5;
  int mb, n; d_lane(mb, n);
  for (int ctl = 0; ctl < 4; ++ctl) {
    const int ct = blockIdx.y * 4 + ctl;                 // column tile of 32
    __syncthreads();
    {                                                    // stage U (64x32) -> UtT (async)
      const int cc = threadIdx.x >> 3, rb = (threadIdx.x & 7) * 8;
      const int gcol = ct * 32 + cc, b = gcol >> 5, c = gcol & 31;
      async_copy16(&UtT[cc * LDU_ + rb],
                   &u16[((size_t)(b * H_ + h)) * L_ + c * T_ + rb]);
    }
    async_wait0();
    __syncthreads();
#pragma unroll
    for (int sub = 0; sub < 2; ++sub) {                  // 16 subtiles / 8 waves
      const int s = w + 8 * sub;
      const int mt = s & 7, nt = s >> 3;                 // P is 128x32 -> 8x2 tiles
      v8f acc = {0.f, 0.f, 0.f, 0.f, 0.f, 0.f, 0.f, 0.f};
#pragma unroll
      for (int kk = 0; kk < 64; kk += 32) {
        v16h a = frag_a(Wl, mt * 16, kk, 64);
        v16h b = frag_bT(UtT, kk, nt * 16, LDU_);
        acc = wmma_f16(a, b, acc);
      }
#pragma unroll
      for (int r = 0; r < 8; ++r) {
        const int j = mt * 16 + mb + r;
        const int gcol = ct * 32 + nt * 16 + n;
        const int b = gcol >> 5, c = gcol & 31;
        S[((size_t)(b * H_ + h) * C_ + c) * 128 + j] = acc[r];
      }
    }
  }
}

// ---------------- sequential scan over chunks (exclusive) ------------------
__global__ void k_scan(const float* __restrict__ wCd, float* __restrict__ S) {
  const int gid = blockIdx.x * 256 + threadIdx.x;        // B*H*N lanes
  const int b = gid / (H_ * N_);
  const int rem = gid % (H_ * N_);
  const int h = rem / N_, n = rem % N_;
  const float ar = wCd[((size_t)h * N_ + n) * 4 + 0];
  const float ai = wCd[((size_t)h * N_ + n) * 4 + 1];
  const float e = expf((float)T_ * ar);                  // w^T
  const float wr = e * cosf((float)T_ * ai), wi = e * sinf((float)T_ * ai);
  float sr = 0.f, si = 0.f;
  float* base = S + ((size_t)(b * H_ + h) * C_) * 128;
  for (int c = 0; c < C_; ++c) {
    const float pr = base[c * 128 + n];
    const float pi = base[c * 128 + N_ + n];
    base[c * 128 + n] = sr;                              // overwrite P w/ state
    base[c * 128 + N_ + n] = si;
    const float nsr = wr * sr - wi * si + pr;
    const float nsi = wr * si + wi * sr + pi;
    sr = nsr; si = nsi;
  }
}

// ------- conv output: Y = Ktri@U + G@S, fused D-skip + GELU -> act16 -------
#define LDS_ 136                                         // ShT row stride
__global__ void k_conv_out(const _Float16* __restrict__ Ktri,
                           const _Float16* __restrict__ G,
                           const float* __restrict__ S,
                           const float* __restrict__ Dv,
                           _Float16* __restrict__ act16) {
  const int h = blockIdx.x;
  __shared__ alignas(16) _Float16 Kl[64 * 64];
  __shared__ alignas(16) _Float16 Gl[64 * 128];
  __shared__ alignas(16) _Float16 UtT[32 * LDU_];        // [col][t]
  __shared__ alignas(16) _Float16 ShT[32 * LDS_];        // [col][j]
#pragma unroll
  for (int i = 0; i < 2; ++i) {                          // Ktri block via async DMA
    const int o = threadIdx.x * 8 + 2048 * i;
    async_copy16(&Kl[o], &Ktri[(size_t)h * 4096 + o]);
  }
#pragma unroll
  for (int i = 0; i < 4; ++i) {                          // G block via async DMA
    const int o = threadIdx.x * 8 + 2048 * i;
    async_copy16(&Gl[o], &G[(size_t)h * 8192 + o]);
  }
  const float Dh = Dv[h];
  async_wait0();
  __syncthreads();
  const int w = threadIdx.x >> 5;
  const int mt = w & 3, nt = w >> 2;                     // 64x32 out -> 4x2 tiles
  int mb, n; d_lane(mb, n);
  for (int ctl = 0; ctl < 4; ++ctl) {
    const int ct = blockIdx.y * 4 + ctl;
    __syncthreads();
    {                                                    // U tile 64x32 -> UtT (async)
      const int cc = threadIdx.x >> 3, rb = (threadIdx.x & 7) * 8;
      const int gcol = ct * 32 + cc, b = gcol >> 5, c = gcol & 31;
      async_copy16(&UtT[cc * LDU_ + rb],
                   &act16[((size_t)(b * H_ + h)) * L_ + c * T_ + rb]);
    }
    {                                                    // S tile 128x32 f32->f16 -> ShT
      const int cc = threadIdx.x >> 3, rb = (threadIdx.x & 7) * 16;
      const int gcol = ct * 32 + cc, b = gcol >> 5, c = gcol & 31;
      const float* sp = &S[((size_t)(b * H_ + h) * C_ + c) * 128 + rb];
      const v4f f0 = *(const v4f*)(sp + 0), f1 = *(const v4f*)(sp + 4);
      const v4f f2 = *(const v4f*)(sp + 8), f3 = *(const v4f*)(sp + 12);
      *(v8h*)&ShT[cc * LDS_ + rb]     = cvt8(f0, f1);
      *(v8h*)&ShT[cc * LDS_ + rb + 8] = cvt8(f2, f3);
    }
    async_wait0();
    __syncthreads();
    v8f acc = {0.f, 0.f, 0.f, 0.f, 0.f, 0.f, 0.f, 0.f};
#pragma unroll
    for (int kk = 0; kk < 64; kk += 32) {                // intra-chunk conv
      v16h a = frag_a(Kl, mt * 16, kk, 64);
      v16h b = frag_bT(UtT, kk, nt * 16, LDU_);
      acc = wmma_f16(a, b, acc);
    }
#pragma unroll
    for (int kk = 0; kk < 128; kk += 32) {               // state -> output
      v16h a = frag_a(Gl, mt * 16, kk, 128);
      v16h b = frag_bT(ShT, kk, nt * 16, LDS_);
      acc = wmma_f16(a, b, acc);
    }
#pragma unroll
    for (int r = 0; r < 8; ++r) {
      const int i = mt * 16 + mb + r;
      const int cc = nt * 16 + n;
      const int gcol = ct * 32 + cc, b = gcol >> 5, c = gcol & 31;
      const float uval = (float)UtT[cc * LDU_ + i];
      const float y = acc[r] + Dh * uval;
      act16[((size_t)(b * H_ + h)) * L_ + c * T_ + i] = (_Float16)gelu_f(y);
    }
  }
}

// ---------------- GLU GEMM: z = Wout @ y + bout ; GLU ; residual -----------
// grid (4, 32, B_), block 256 (8 waves).
#define LDB_ 40                                          // BtT row stride
__global__ void k_glu_gemm(const _Float16* __restrict__ y16,
                           const float* __restrict__ Wout,
                           const float* __restrict__ bout,
                           float* __restrict__ Hres) {
  const int m0 = blockIdx.x * 64;
  const int l0 = blockIdx.y * 64;
  const int b  = blockIdx.z;
  __shared__ alignas(16) _Float16 A1[64 * 32];
  __shared__ alignas(16) _Float16 A2[64 * 32];
  __shared__ alignas(16) _Float16 BtT[64 * LDB_];        // [l][k]
  const int w = threadIdx.x >> 5;
  int mb, n; d_lane(mb, n);
  v8f acc1[2], acc2[2];
#pragma unroll
  for (int s = 0; s < 2; ++s) {
    acc1[s] = (v8f){0.f, 0.f, 0.f, 0.f, 0.f, 0.f, 0.f, 0.f};
    acc2[s] = (v8f){0.f, 0.f, 0.f, 0.f, 0.f, 0.f, 0.f, 0.f};
  }
  for (int kb = 0; kb < H_; kb += 32) {
    __syncthreads();
    {                                                    // A tiles 64x32, f32 -> f16
      const int row = threadIdx.x >> 2, kk = (threadIdx.x & 3) * 8;
      const float* g1 = &Wout[(size_t)(m0 + row) * H_ + kb + kk];
      const float* g2 = &Wout[(size_t)(m0 + 256 + row) * H_ + kb + kk];
      *(v8h*)&A1[row * 32 + kk] = cvt8(*(const v4f*)g1, *(const v4f*)(g1 + 4));
      *(v8h*)&A2[row * 32 + kk] = cvt8(*(const v4f*)g2, *(const v4f*)(g2 + 4));
    }
    {                                                    // B tile 32x64 -> BtT [l][k]
      const int row = threadIdx.x >> 3, cb = (threadIdx.x & 7) * 8;
      const _Float16* src = &y16[((size_t)(b * H_ + kb + row)) * L_ + l0 + cb];
      const v8h d = *(const v8h*)src;
      __builtin_prefetch(src + 32 * L_, 0, 0);           // next K-step stream
#pragma unroll
      for (int e = 0; e < 8; ++e) BtT[(cb + e) * LDB_ + row] = d[e];
    }
    __syncthreads();
#pragma unroll
    for (int s = 0; s < 2; ++s) {                        // 16 subtiles / 8 waves
      const int st = w + 8 * s;
      const int mt = st & 3, nt = st >> 2;               // 4 x 4 grid of 16x16
      v16h bf = frag_bT(BtT, 0, nt * 16, LDB_);
      v16h a1 = frag_a(A1, mt * 16, 0, 32);
      acc1[s] = wmma_f16(a1, bf, acc1[s]);
      v16h a2 = frag_a(A2, mt * 16, 0, 32);
      acc2[s] = wmma_f16(a2, bf, acc2[s]);
    }
  }
#pragma unroll
  for (int s = 0; s < 2; ++s) {
    const int st = w + 8 * s;
    const int mt = st & 3, nt = st >> 2;
#pragma unroll
    for (int r = 0; r < 8; ++r) {
      const int o = m0 + mt * 16 + mb + r;               // z1 row (== output h)
      const int l = l0 + nt * 16 + n;
      const float z1 = acc1[s][r] + bout[o];
      const float z2 = acc2[s][r] + bout[o + H_];
      const float g = z1 * sigmoid_f(z2);
      float* p = Hres + ((size_t)(b * H_ + o)) * L_ + l;
      *p = *p + g;                                       // residual
    }
  }
}

// ---------------------------------------------------------------------------
extern "C" void kernel_launch(void* const* d_in, const int* in_sizes, int n_in,
                              void* d_out, int out_size, void* d_ws, size_t ws_size,
                              hipStream_t stream) {
  (void)in_sizes; (void)n_in; (void)out_size; (void)ws_size;
  const float* x       = (const float*)d_in[0];
  const float* log_dt  = (const float*)d_in[1];
  const float* Alog    = (const float*)d_in[2];
  const float* Aim     = (const float*)d_in[3];
  const float* Cre     = (const float*)d_in[4];
  const float* Cim     = (const float*)d_in[5];
  const float* Dv      = (const float*)d_in[6];
  const float* Wout    = (const float*)d_in[7];
  const float* bout    = (const float*)d_in[8];
  const float* lns     = (const float*)d_in[9];
  const float* lnb     = (const float*)d_in[10];
  float* out           = (float*)d_out;

  // workspace carve-up (256B aligned)
  char* ws = (char*)d_ws;
  const size_t SZ_ACT  = (size_t)B_ * H_ * L_ * sizeof(_Float16);
  const size_t SZ_HRES = (size_t)B_ * H_ * L_ * sizeof(float);
  const size_t SZ_S    = (size_t)B_ * H_ * C_ * 128 * sizeof(float);
  const size_t SZ_KTRI = (size_t)H_ * 64 * 64 * sizeof(_Float16);
  const size_t SZ_WST  = (size_t)H_ * 128 * 64 * sizeof(_Float16);
  const size_t SZ_G    = (size_t)H_ * 64 * 128 * sizeof(_Float16);
  size_t off = 0;
  _Float16* act16 = (_Float16*)(ws + off); off += (SZ_ACT  + 255) & ~(size_t)255;
  float*    Hres  = (float*)   (ws + off); off += (SZ_HRES + 255) & ~(size_t)255;
  float*    S     = (float*)   (ws + off); off += (SZ_S    + 255) & ~(size_t)255;
  _Float16* Ktri  = (_Float16*)(ws + off); off += (SZ_KTRI + 255) & ~(size_t)255;
  _Float16* Wst   = (_Float16*)(ws + off); off += (SZ_WST  + 255) & ~(size_t)255;
  _Float16* G     = (_Float16*)(ws + off); off += (SZ_G    + 255) & ~(size_t)255;
  float*    wCd   = (float*)   (ws + off);

  k_tr_LH_to_HL<<<dim3(H_ / 32, L_ / 32, B_), dim3(32, 8), 0, stream>>>(x, Hres);

  for (int i = 0; i < NL_; ++i) {
    const float* ldt = log_dt + (size_t)i * H_;
    const float* Al  = Alog   + (size_t)i * H_ * N_;
    const float* Ai  = Aim    + (size_t)i * H_ * N_;
    const float* Cr  = Cre    + (size_t)i * H_ * N_;
    const float* Ci  = Cim    + (size_t)i * H_ * N_;
    const float* Dl  = Dv     + (size_t)i * H_;
    const float* Wl  = Wout   + (size_t)i * H2_ * H_;
    const float* bl  = bout   + (size_t)i * H2_;
    const float* sl  = lns    + (size_t)i * H_;
    const float* bnl = lnb    + (size_t)i * H_;

    k_layernorm<<<(B_ * L_) / 256, 256, 0, stream>>>(Hres, sl, bnl, act16);
    k_build<<<H_, T_, 0, stream>>>(ldt, Al, Ai, Cr, Ci, wCd, Ktri, Wst, G);
    k_chunk_states<<<dim3(H_, 4), 256, 0, stream>>>(act16, Wst, S);
    k_scan<<<(B_ * H_ * N_) / 256, 256, 0, stream>>>(wCd, S);
    k_conv_out<<<dim3(H_, 4), 256, 0, stream>>>(Ktri, G, S, Dl, act16);
    k_glu_gemm<<<dim3(4, L_ / 64, B_), 256, 0, stream>>>(act16, Wl, bl, Hres);
  }

  k_tr_HL_to_LH<<<dim3(L_ / 32, H_ / 32, B_), dim3(32, 8), 0, stream>>>(Hres, out);
}
